// layer_24481313587862
// MI455X (gfx1250) — matrix operand order repaired and checked by reading, not compiled, not run on hardware
//
#include <hip/hip_runtime.h>

// ---------------------------------------------------------------------------
// MI455X (gfx1250, wave32) fused RBF-MoE layer.
// B=8, S=4096, D=512, E=64, H=512.
// All three contractions run on v_wmma_f32_16x16x32_f16 with a 2-term f16
// split (hi/lo) per operand for ~fp32 accuracy: x*y ~= xh*yh + xh*yl + xl*yh,
// accumulated in f32 by the WMMA unit. Memory-roofline bound (~200MB @
// 23.3 TB/s ~ 9us). Centers are pre-split once (kills redundant cvt VALU);
// the S-contraction is K-split 4x for WGP occupancy; all reductions are
// fixed-order (no float atomics) so replays are bit-deterministic.
// ---------------------------------------------------------------------------

typedef _Float16 f16;
typedef __attribute__((ext_vector_type(16))) _Float16 v16h;
typedef __attribute__((ext_vector_type(8)))  _Float16 v8h;
typedef __attribute__((ext_vector_type(8)))  float    v8f;

static constexpr int Bn = 8;
static constexpr int Ss = 4096;
static constexpr int Dd = 512;
static constexpr int Ee = 64;
static constexpr int Hh = 512;
static constexpr int KSPLIT = 4;              // S-chunks in k_hp
static constexpr int SCHUNK = Ss / KSPLIT;    // 1024

static __device__ __forceinline__ v8f wmma16(v16h a, v16h b, v8f c) {
  // (neg_a, A, neg_b, B, c_mod, C, reuse_a, reuse_b)
  return __builtin_amdgcn_wmma_f32_16x16x32_f16(false, a, false, b, (short)0, c,
                                                false, false);
}

static __device__ __forceinline__ v16h cat8(v8h a, v8h b) {
  return __builtin_shufflevector(a, b, 0, 1, 2, 3, 4, 5, 6, 7, 8, 9, 10, 11,
                                 12, 13, 14, 15);
}

// Split f32 into hi/lo f16 pair, writing element j of the two fragment vectors.
static __device__ __forceinline__ void split_into(float x, v16h& vhi, v16h& vlo,
                                                  int j) {
  f16 h = (f16)x;
  vhi[j] = h;
  vlo[j] = (f16)(x - (float)h);
}

static __device__ __forceinline__ void split32(float x, f16& hi, f16& lo) {
  f16 h = (f16)x;
  hi = h;
  lo = (f16)(x - (float)h);
}

// A-fragment k index (16-bit A, 16x32): lane L, element j
//   k = 8*(L>>4) + (j&7) + 16*(j>>3)
static __device__ __forceinline__ int a_k(int j) {
  return (j & 7) + ((j >> 3) << 4);
}

// --------------------------------------------------------------------------
// Kernel 0: c2[e] = ||centers[e]||^2 ; pre-split centers into f16 hi/lo.
// --------------------------------------------------------------------------
__global__ __launch_bounds__(256) void k_init(const float* __restrict__ cen,
                                              float* __restrict__ c2,
                                              f16* __restrict__ chi,
                                              f16* __restrict__ clo) {
  int tid = threadIdx.x;
  if (tid < Ee) {
    const float* row = cen + (size_t)tid * Dd;
    float s = 0.f;
    for (int d = 0; d < Dd; ++d) { float v = row[d]; s += v * v; }
    c2[tid] = s;
  }
  for (int i = tid; i < Ee * Dd; i += 256) {
    f16 hi, lo; split32(cen[i], hi, lo);
    chi[i] = hi;
    clo[i] = lo;
  }
}

// --------------------------------------------------------------------------
// Kernel 1: logits tile (64 rows x 64 experts) = h_tile @ centers^T via WMMA,
// softmax over E in LDS, emit prob transposed as f16 hi/lo [B][E][S].
// Grid = B*S/64 = 512 blocks, 256 thr.
// --------------------------------------------------------------------------
__global__ __launch_bounds__(256) void k_prob(const float* __restrict__ hsrc,
                                              const f16* __restrict__ chi,
                                              const f16* __restrict__ clo,
                                              const float* __restrict__ c2g,
                                              f16* __restrict__ phi,
                                              f16* __restrict__ plo) {
  __shared__ float lds[64 * 65];   // stride 65: conflict-free rows & cols
  __shared__ float c2s[Ee];

  const int tid  = threadIdx.x;
  const int wave = tid >> 5;
  const int lane = tid & 31;
  const int b    = blockIdx.x >> 6;          // 64 blocks per batch
  const int s0   = (blockIdx.x & 63) << 6;   // 64 rows of S

  if (tid < Ee) c2s[tid] = c2g[tid];

  const int rt   = wave >> 1;                // row tile 0..3
  const int c0t  = (wave & 1) * 2;           // col tiles c0t, c0t+1
  const int m    = lane & 15;
  const int hi16 = lane >> 4;

  const float* hrow = hsrc + ((size_t)(b * Ss + s0 + rt * 16 + m)) * Dd;
  // B-fragment rows (experts) for the two column tiles: pre-split f16.
  const v8h* ch0 = (const v8h*)(chi + (size_t)((c0t + 0) * 16 + m) * Dd);
  const v8h* cl0 = (const v8h*)(clo + (size_t)((c0t + 0) * 16 + m) * Dd);
  const v8h* ch1 = (const v8h*)(chi + (size_t)((c0t + 1) * 16 + m) * Dd);
  const v8h* cl1 = (const v8h*)(clo + (size_t)((c0t + 1) * 16 + m) * Dd);

  v8f acc0 = {}, acc1 = {};

  for (int ks = 0; ks < Dd; ks += 32) {
    v16h a_hi, a_lo;
    const int kb = ks + 16 * hi16;
#pragma unroll
    for (int j = 0; j < 16; ++j) split_into(hrow[kb + j], a_hi, a_lo, j);

    const int c8 = (ks + 16 * hi16) >> 3;    // 16 consecutive halves = 2x v8h
    v16h b0h = cat8(ch0[c8], ch0[c8 + 1]);
    v16h b0l = cat8(cl0[c8], cl0[c8 + 1]);
    v16h b1h = cat8(ch1[c8], ch1[c8 + 1]);
    v16h b1l = cat8(cl1[c8], cl1[c8 + 1]);

    acc0 = wmma16(a_hi, b0h, acc0);
    acc0 = wmma16(a_hi, b0l, acc0);
    acc0 = wmma16(a_lo, b0h, acc0);
    acc1 = wmma16(a_hi, b1h, acc1);
    acc1 = wmma16(a_hi, b1l, acc1);
    acc1 = wmma16(a_lo, b1h, acc1);
  }

  // C-fragment -> LDS: row = rt*16 + v + 8*hi16, col = ct*16 + (lane&15)
#pragma unroll
  for (int v = 0; v < 8; ++v) {
    int r = rt * 16 + v + 8 * hi16;
    lds[r * 65 + c0t * 16 + m]        = acc0[v];
    lds[r * 65 + (c0t + 1) * 16 + m]  = acc1[v];
  }
  __syncthreads();

  if (tid < 64) {
    // logit = hc - c2/2  (h^2 term cancels in softmax)
    float mx = -1e30f;
#pragma unroll 4
    for (int e = 0; e < Ee; ++e) {
      float l = lds[tid * 65 + e] - 0.5f * c2s[e];
      lds[tid * 65 + e] = l;
      mx = fmaxf(mx, l);
    }
    float sum = 0.f;
#pragma unroll 4
    for (int e = 0; e < Ee; ++e) {
      float p = __expf(lds[tid * 65 + e] - mx);
      lds[tid * 65 + e] = p;
      sum += p;
    }
    float inv = 1.f / sum;
    const size_t sidx = (size_t)s0 + tid;
#pragma unroll 4
    for (int e = 0; e < Ee; ++e) {
      float p = lds[tid * 65 + e] * inv;
      f16 hi, lo; split32(p, hi, lo);
      size_t off = ((size_t)b * Ee + e) * Ss + sidx;  // transposed layout
      phi[off] = hi;
      plo[off] = lo;
    }
  }
}

// --------------------------------------------------------------------------
// Kernel 2: ps[b][e] = sum_s prob[b][s][e]. One block per (b,e), fixed-order
// tree reduction -> bit-deterministic (no float atomics). Grid = 512 blocks.
// --------------------------------------------------------------------------
__global__ __launch_bounds__(256) void k_ps(const f16* __restrict__ phi,
                                            const f16* __restrict__ plo,
                                            float* __restrict__ ps) {
  __shared__ float red[256];
  const int tid = threadIdx.x;
  const int be  = blockIdx.x;
  const f16* ph = phi + (size_t)be * Ss;
  const f16* pl = plo + (size_t)be * Ss;
  float s = 0.f;
  for (int i = tid; i < Ss; i += 256) s += (float)ph[i] + (float)pl[i];
  red[tid] = s;
  __syncthreads();
#pragma unroll
  for (int off = 128; off > 0; off >>= 1) {
    if (tid < off) red[tid] += red[tid + off];
    __syncthreads();
  }
  if (tid == 0) ps[be] = red[0];
}

// --------------------------------------------------------------------------
// Kernel 3: hp_part[c][b][e][d] = sum_{s in chunk c} prob[b][s][e]*h[b][s][d].
// M=E=64, N=64-wide d-tile, K=SCHUNK=1024. Grid = KSPLIT*B*(D/64) = 256
// blocks, 256 thr — 4x the occupancy of an unsplit S-contraction.
// --------------------------------------------------------------------------
__global__ __launch_bounds__(256) void k_hp(const float* __restrict__ hsrc,
                                            const f16* __restrict__ phi,
                                            const f16* __restrict__ plo,
                                            float* __restrict__ hp_part) {
  const int tid   = threadIdx.x;
  const int wave  = tid >> 5;
  const int lane  = tid & 31;
  const int chunk = blockIdx.x >> 6;         // 0..KSPLIT-1
  const int b     = (blockIdx.x >> 3) & 7;
  const int d0    = (blockIdx.x & 7) * 64;

  const int rt   = wave >> 1;                // e tile 0..3
  const int c0t  = (wave & 1) * 2;           // d tiles c0t, c0t+1
  const int m    = lane & 15;
  const int hi16 = lane >> 4;

  const int e = rt * 16 + m;
  const v8h* ph = (const v8h*)(phi + ((size_t)b * Ee + e) * Ss);
  const v8h* pl = (const v8h*)(plo + ((size_t)b * Ee + e) * Ss);

  v8f acc0 = {}, acc1 = {};

  const int ks0 = chunk * SCHUNK;
  for (int ks = ks0; ks < ks0 + SCHUNK; ks += 32) {
    const int c8 = (ks + 8 * hi16) >> 3;     // element j: k = kb + a_k(j)
    v16h a_hi = cat8(ph[c8], ph[c8 + 2]);    // halves kb..kb+7, kb+16..kb+23
    v16h a_lo = cat8(pl[c8], pl[c8 + 2]);

#pragma unroll
    for (int t = 0; t < 2; ++t) {
      const int d = d0 + (c0t + t) * 16 + m;
      const float* hcol = hsrc + ((size_t)b * Ss + ks + 8 * hi16) * Dd + d;
      v16h b_hi, b_lo;
#pragma unroll
      for (int j = 0; j < 16; ++j)
        split_into(hcol[(size_t)a_k(j) * Dd], b_hi, b_lo, j);
      v8f& acc = t ? acc1 : acc0;
      acc = wmma16(a_hi, b_hi, acc);
      acc = wmma16(a_hi, b_lo, acc);
      acc = wmma16(a_lo, b_hi, acc);
    }
  }

  float* dst = hp_part + (size_t)chunk * Bn * Ee * Dd;
#pragma unroll
  for (int v = 0; v < 8; ++v) {
    int er = rt * 16 + v + 8 * hi16;
    dst[((size_t)b * Ee + er) * Dd + d0 + c0t * 16 + m]       = acc0[v];
    dst[((size_t)b * Ee + er) * Dd + d0 + (c0t + 1) * 16 + m] = acc1[v];
  }
}

// --------------------------------------------------------------------------
// Kernel 4: hp = sum of KSPLIT partials (fixed order -> deterministic).
// --------------------------------------------------------------------------
__global__ __launch_bounds__(256) void k_red(const float* __restrict__ hp_part,
                                             float* __restrict__ hp) {
  const int n = Bn * Ee * Dd;                // 262144
  int i = blockIdx.x * 1024 + threadIdx.x;
#pragma unroll
  for (int r = 0; r < 4; ++r, i += 256) {
    float s = 0.f;
#pragma unroll
    for (int c = 0; c < KSPLIT; ++c) s += hp_part[(size_t)c * n + i];
    hp[i] = s;
  }
}

// --------------------------------------------------------------------------
// Kernel 5: out[b][e][h] = sum_d hp[b][e][d]*W[e][h][d] + ps[b][e]*bias[e][h].
// Per expert: M=16 (8 valid batch rows), N=64-wide h-tile, K=D=512.
// Grid = E*(H/64) = 512 blocks, 128 threads. Streams W exactly once (64 MiB).
// --------------------------------------------------------------------------
__global__ __launch_bounds__(128) void k_out(const float* __restrict__ hp,
                                             const float* __restrict__ Wt,
                                             const float* __restrict__ bias,
                                             const float* __restrict__ ps,
                                             float* __restrict__ out) {
  const int tid  = threadIdx.x;
  const int ct   = tid >> 5;                 // n tile 0..3
  const int lane = tid & 31;
  const int e    = blockIdx.x >> 3;
  const int h0   = (blockIdx.x & 7) * 64;

  const int m    = lane & 15;                // A: batch row / B,C: column
  const int hi16 = lane >> 4;

  v8f acc = {};

  for (int ks = 0; ks < Dd; ks += 32) {
    v16h a_hi = {}, a_lo = {};
    if (m < 8) {                             // batch rows 8..15 are zero-pad
      const float* hprow = hp + ((size_t)m * Ee + e) * Dd;
      const int kb = ks + 8 * hi16;
#pragma unroll
      for (int j = 0; j < 16; ++j)
        split_into(hprow[kb + a_k(j)], a_hi, a_lo, j);
    }
    const int hn = h0 + ct * 16 + m;
    const float* wrow = Wt + ((size_t)e * Hh + hn) * Dd + ks + 8 * hi16;
    v16h b_hi, b_lo;
#pragma unroll
    for (int j = 0; j < 16; ++j) split_into(wrow[a_k(j)], b_hi, b_lo, j);

    acc = wmma16(a_hi, b_hi, acc);           // full EXEC here (reconverged)
    acc = wmma16(a_hi, b_lo, acc);
    acc = wmma16(a_lo, b_hi, acc);
  }

  // C layout: lane n = m, row = v + 8*hi16 -> batches 0..7 live in lanes 0..15
  if (hi16 == 0) {
#pragma unroll
    for (int v = 0; v < 8; ++v) {
      int hn = h0 + ct * 16 + m;
      out[((size_t)v * Ee + e) * Hh + hn] =
          acc[v] + ps[v * Ee + e] * bias[(size_t)e * Hh + hn];
    }
  }
}

// --------------------------------------------------------------------------
extern "C" void kernel_launch(void* const* d_in, const int* in_sizes, int n_in,
                              void* d_out, int out_size, void* d_ws,
                              size_t ws_size, hipStream_t stream) {
  (void)in_sizes; (void)n_in; (void)out_size; (void)ws_size;
  const float* h       = (const float*)d_in[0];   // [B,S,D]
  const float* centers = (const float*)d_in[1];   // [E,D]
  const float* W       = (const float*)d_in[2];   // [E,H,D]
  const float* bias    = (const float*)d_in[3];   // [E,H]
  float* out = (float*)d_out;                     // [B,E,H]

  char* ws = (char*)d_ws;
  size_t off = 0;
  float* c2 = (float*)(ws + off); off += 4096;                    // 256 B used
  float* ps = (float*)(ws + off); off += 4096;                    // 2 KiB used
  f16* chi  = (f16*)(ws + off);   off += (size_t)Ee * Dd * 2;     // 64 KiB
  f16* clo  = (f16*)(ws + off);   off += (size_t)Ee * Dd * 2;     // 64 KiB
  float* hp = (float*)(ws + off); off += (size_t)Bn * Ee * Dd * 4;        // 1M
  float* hpp = (float*)(ws + off); off += (size_t)KSPLIT * Bn * Ee * Dd * 4;
  f16* phi  = (f16*)(ws + off);   off += (size_t)Bn * Ee * Ss * 2;        // 4M
  f16* plo  = (f16*)(ws + off);   off += (size_t)Bn * Ee * Ss * 2;        // 4M

  k_init<<<1, 256, 0, stream>>>(centers, c2, chi, clo);
  k_prob<<<Bn * Ss / 64, 256, 0, stream>>>(h, chi, clo, c2, phi, plo);
  k_ps<<<Bn * Ee, 256, 0, stream>>>(phi, plo, ps);
  k_hp<<<KSPLIT * Bn * (Dd / 64), 256, 0, stream>>>(h, phi, plo, hpp);
  k_red<<<Bn * Ee * Dd / 1024, 256, 0, stream>>>(hpp, hp);
  k_out<<<Ee * (Hh / 64), 128, 0, stream>>>(hp, W, bias, ps, out);
}
